// AtomGNN_1460288881373
// MI455X (gfx1250) — matrix operand order repaired and checked by reading, not compiled, not run on hardware
//
#include <hip/hip_runtime.h>
#include <hip/hip_bf16.h>
#include <cstdint>
#include <cstddef>

typedef __attribute__((ext_vector_type(16))) __bf16 v16bf;
typedef __attribute__((ext_vector_type(8)))  float  v8f;

// ---------------------------------------------------------------------------
// Weight convert + pad: fp32 [rows][kin] -> bf16 [rowsPad][kpad] (zero pad)
// ---------------------------------------------------------------------------
__global__ void cvt_pad_kernel(const float* __restrict__ S, __bf16* __restrict__ Dp,
                               int rows, int kin, int rowsPad, int kpad)
{
    int idx = blockIdx.x * blockDim.x + threadIdx.x;
    int total = rowsPad * kpad;
    if (idx >= total) return;
    int r = idx / kpad;
    int k = idx - r * kpad;
    float v = (r < rows && k < kin) ? S[r * kin + k] : 0.0f;
    Dp[idx] = (__bf16)v;
}

// ---------------------------------------------------------------------------
// GEMM: Y[M][Nreal] = act( (X (+Xadd)) [M][K] @ Wb[Npad][K]^T + bias )
// One wave per 16-row M tile; NT = Npad/16 column tiles, KSTEPS = K/32.
// Fully unrolled K loop -> NT*KSTEPS WMMAs per wave. All indexing is 32-bit
// element offsets so the compiler can use GVS (saddr + i32 voffset) addressing.
// ---------------------------------------------------------------------------
template<int NT, int KSTEPS>
__global__ void gemm_bf16_kernel(const float* __restrict__ X, const float* __restrict__ Xadd,
                                 const __bf16* __restrict__ Wb, const float* __restrict__ bias,
                                 float* __restrict__ Y, int M, int Nreal, int relu)
{
    constexpr int K = KSTEPS * 32;
    int lane = threadIdx.x & 31;
    int wid  = threadIdx.x >> 5;
    int mtile = blockIdx.x * (blockDim.x >> 5) + wid;
    if (mtile * 16 >= M) return;
    int m0 = mtile * 16;
    int le = lane & 15;
    int hl = lane >> 4;

    v8f zero = {};
    v8f acc[NT];
#pragma unroll
    for (int t = 0; t < NT; ++t) acc[t] = zero;

    const int rowoff = (m0 + le) * K;           // 32-bit element offset
    __builtin_prefetch(X + rowoff + (int)(blockDim.x >> 5) * 16 * K, 0, 3);

#pragma unroll
    for (int ks = 0; ks < KSTEPS; ++ks) {
        const int k0 = ks * 32;
        v16bf a;
#pragma unroll
        for (int j = 0; j < 8; ++j) {
            int o0 = rowoff + k0 + 16 * hl + j;
            int o1 = o0 + 8;
            float f0 = X[o0];
            float f1 = X[o1];
            if (Xadd) {
                f0 += Xadd[o0];
                f1 += Xadd[o1];
            }
            a[j]     = (__bf16)f0;
            a[8 + j] = (__bf16)f1;
        }
#pragma unroll
        for (int t = 0; t < NT; ++t) {
            v16bf b = *(const v16bf*)(Wb + (t * 16 + le) * K + (k0 + 16 * hl));
            acc[t] = __builtin_amdgcn_wmma_f32_16x16x32_bf16(
                        false, a, false, b, (short)0, acc[t], false, false);
        }
    }

#pragma unroll
    for (int t = 0; t < NT; ++t) {
        int n = t * 16 + le;
        float bval = (n < Nreal) ? bias[n] : 0.0f;
#pragma unroll
        for (int r = 0; r < 8; ++r) {
            int m = m0 + r + 8 * hl;
            float v = acc[t][r] + bval;
            if (relu) v = fmaxf(v, 0.0f);
            if (n < Nreal) Y[m * Nreal + n] = v;
        }
    }
}

// ---------------------------------------------------------------------------
// Fused GINE edge pass for a 16-edge tile per wave (D = NT*16 = 64 or 128):
//   e      = (edge_attr * w) @ We^T + be        (WMMA, K=16 zero-padded to 32)
//   msg    = relu(H[src] + e)
//   Agg[dst] += msg                              (f32 global atomics)
// Gathers for all 8 rows are issued before any consumer (one loadcnt wait);
// all offsets are 32-bit element indices (GVS addressing + scale_offset).
// ---------------------------------------------------------------------------
template<int NT>
__global__ void edge_msg_kernel(const float* __restrict__ eattr, const float* __restrict__ ew,
                                const int* __restrict__ srcI, const int* __restrict__ dstI,
                                const __bf16* __restrict__ Web /*[NT*16][32]*/,
                                const float* __restrict__ be,
                                const float* __restrict__ H, float* __restrict__ Agg, int E)
{
    const int D = NT * 16;
    int lane = threadIdx.x & 31;
    int wid  = threadIdx.x >> 5;
    int etile = blockIdx.x * (blockDim.x >> 5) + wid;
    int e0 = etile * 16;
    if (e0 >= E) return;
    int le = lane & 15;
    int hl = lane >> 4;

    // A fragment: row m = edge e0+le, real K = 16, upper half zero.
    int em = e0 + le;
    float w = ew[em];
    int eoff = em * 16;
    v16bf a;
#pragma unroll
    for (int j = 0; j < 8; ++j) {
        a[j]     = hl ? (__bf16)0.0f : (__bf16)(eattr[eoff + j] * w);
        a[8 + j] = hl ? (__bf16)0.0f : (__bf16)(eattr[eoff + 8 + j] * w);
    }

    int hoffs[8], doffs[8];
#pragma unroll
    for (int r = 0; r < 8; ++r) {
        int e = e0 + r + 8 * hl;
        hoffs[r] = srcI[e] * D;                 // 32-bit row base offsets
        doffs[r] = dstI[e] * D;
    }
    float bvals[NT];
#pragma unroll
    for (int t = 0; t < NT; ++t) bvals[t] = be[t * 16 + le];

    v8f zero = {};
#pragma unroll
    for (int t = 0; t < NT; ++t) {
        v16bf b = *(const v16bf*)(Web + (t * 16 + le) * 32 + 16 * hl);
        v8f acc = __builtin_amdgcn_wmma_f32_16x16x32_bf16(
                     false, a, false, b, (short)0, zero, false, false);
        int n = t * 16 + le;

        float hv[8];
#pragma unroll
        for (int r = 0; r < 8; ++r) hv[r] = H[hoffs[r] + n];   // batch gathers
#pragma unroll
        for (int r = 0; r < 8; ++r) {
            float v = fmaxf(acc[r] + bvals[t] + hv[r], 0.0f);
            atomicAdd(&Agg[doffs[r] + n], v);
        }
    }
}

// ---------------------------------------------------------------------------
// Host-side orchestration
// ---------------------------------------------------------------------------
extern "C" void kernel_launch(void* const* d_in, const int* in_sizes, int n_in,
                              void* d_out, int out_size, void* d_ws, size_t ws_size,
                              hipStream_t stream)
{
    const float* x   = (const float*)d_in[0];
    const int*   ei  = (const int*)d_in[1];
    const float* eat = (const float*)d_in[2];
    const float* ew  = (const float*)d_in[3];
    const float* We1 = (const float*)d_in[4];
    const float* be1 = (const float*)d_in[5];
    const float* W11 = (const float*)d_in[6];
    const float* b11 = (const float*)d_in[7];
    const float* W12 = (const float*)d_in[8];
    const float* b12 = (const float*)d_in[9];
    const float* We2 = (const float*)d_in[10];
    const float* be2 = (const float*)d_in[11];
    const float* W21 = (const float*)d_in[12];
    const float* b21 = (const float*)d_in[13];
    const float* W22 = (const float*)d_in[14];
    const float* b22 = (const float*)d_in[15];
    const float* Wo  = (const float*)d_in[16];
    const float* bo  = (const float*)d_in[17];
    float* out = (float*)d_out;

    const int N = 50000;
    const int E = 800000;
    const int* src = ei;
    const int* dst = ei + E;

    // workspace carve-out (256B aligned)
    char* ws = (char*)d_ws;
    size_t off = 0;
    auto alloc = [&](size_t bytes) -> void* {
        off = (off + 255) & ~(size_t)255;
        void* p = ws + off;
        off += bytes;
        return p;
    };
    float*  agg1 = (float*) alloc((size_t)N * 64  * sizeof(float));
    float*  tbuf = (float*) alloc((size_t)N * 128 * sizeof(float));
    float*  hmid = (float*) alloc((size_t)N * 128 * sizeof(float));
    float*  agg2 = (float*) alloc((size_t)N * 128 * sizeof(float));
    float*  hfin = (float*) alloc((size_t)N * 128 * sizeof(float));
    __bf16* We1b = (__bf16*)alloc((size_t)64  * 32  * 2);
    __bf16* We2b = (__bf16*)alloc((size_t)128 * 32  * 2);
    __bf16* W11b = (__bf16*)alloc((size_t)128 * 64  * 2);
    __bf16* W12b = (__bf16*)alloc((size_t)128 * 128 * 2);
    __bf16* W21b = (__bf16*)alloc((size_t)128 * 128 * 2);
    __bf16* W22b = (__bf16*)alloc((size_t)128 * 128 * 2);
    __bf16* Wob  = (__bf16*)alloc((size_t)112 * 128 * 2);

    auto cvt = [&](const float* S, __bf16* Dp, int rows, int kin, int rowsPad, int kpad) {
        int total = rowsPad * kpad;
        cvt_pad_kernel<<<(total + 255) / 256, 256, 0, stream>>>(S, Dp, rows, kin, rowsPad, kpad);
    };
    cvt(We1, We1b, 64, 16, 64, 32);
    cvt(We2, We2b, 128, 16, 128, 32);
    cvt(W11, W11b, 128, 64, 128, 64);
    cvt(W12, W12b, 128, 128, 128, 128);
    cvt(W21, W21b, 128, 128, 128, 128);
    cvt(W22, W22b, 128, 128, 128, 128);
    cvt(Wo,  Wob,  100, 128, 112, 128);

    hipMemsetAsync(agg1, 0, (size_t)N * 64 * sizeof(float), stream);
    hipMemsetAsync(agg2, 0, (size_t)N * 128 * sizeof(float), stream);

    const int etiles = E / 16;                 // 50000, exact
    const int eblocks = (etiles + 3) / 4;      // 4 waves / block
    const int mtiles = N / 16;                 // 3125, exact
    const int gblocks = (mtiles + 3) / 4;

    // Layer 1: edge messages (D=64) into agg1, then node MLP (residual fused)
    edge_msg_kernel<4><<<eblocks, 128, 0, stream>>>(eat, ew, src, dst, We1b, be1, x, agg1, E);
    gemm_bf16_kernel<8, 2><<<gblocks, 128, 0, stream>>>(x,    agg1,    W11b, b11, tbuf, N, 128, 1);
    gemm_bf16_kernel<8, 4><<<gblocks, 128, 0, stream>>>(tbuf, nullptr, W12b, b12, hmid, N, 128, 1);

    // Layer 2: edge messages (D=128) into agg2, then node MLP (residual fused)
    edge_msg_kernel<8><<<eblocks, 128, 0, stream>>>(eat, ew, src, dst, We2b, be2, hmid, agg2, E);
    gemm_bf16_kernel<8, 4><<<gblocks, 128, 0, stream>>>(hmid, agg2,    W21b, b21, tbuf, N, 128, 1);
    gemm_bf16_kernel<8, 4><<<gblocks, 128, 0, stream>>>(tbuf, nullptr, W22b, b22, hfin, N, 128, 1);

    // Output head: [N,128] @ [128,100] (padded to 112 cols, masked store)
    gemm_bf16_kernel<7, 4><<<gblocks, 128, 0, stream>>>(hfin, nullptr, Wob, bo, out, N, 100, 0);
}
